// DecoderTransformer_11785390260376
// MI455X (gfx1250) — matrix operand order repaired and checked by reading, not compiled, not run on hardware
//
#include <hip/hip_runtime.h>
#include <hip/hip_bf16.h>
#include <math.h>

// Model constants (reference: V,T,E,H,L = 32000,1024,1024,16,6 ; B=2 ; D=64 ; F=4096)
#define V_ 32000
#define T_ 1024
#define E_ 1024
#define H_ 16
#define L_ 6
#define B_ 2
#define D_ 64
#define F_ 4096

typedef __bf16 bf16;
typedef __bf16  v16bf __attribute__((ext_vector_type(16)));
typedef __bf16  v8bf  __attribute__((ext_vector_type(8)));
typedef float   v8f   __attribute__((ext_vector_type(8)));
typedef int     v4i   __attribute__((ext_vector_type(4)));

#if __has_builtin(__builtin_amdgcn_global_load_async_to_lds_b128)
#define HAS_ASYNC_LDS 1
typedef __attribute__((address_space(3))) v4i lds_v4i;
#else
#define HAS_ASYNC_LDS 0
#endif

__device__ __forceinline__ bf16 f2bf(float f) {
  unsigned int u = __float_as_uint(f);
  u += 0x7FFFu + ((u >> 16) & 1u);           // round-to-nearest-even
  unsigned short h = (unsigned short)(u >> 16);
  return __builtin_bit_cast(__bf16, h);
}

// Two aligned 8x bf16 (128-bit) loads -> one 16-wide fragment register block.
__device__ __forceinline__ v16bf ldfrag(const bf16* p0, const bf16* p1) {
  v8bf a = *(const v8bf*)p0;
  v8bf b = *(const v8bf*)p1;
  return __builtin_shufflevector(a, b, 0,1,2,3,4,5,6,7,8,9,10,11,12,13,14,15);
}

// Copy 32 bytes global -> LDS. Prefer CDNA5 async DMA (ASYNCcnt) when available.
__device__ __forceinline__ void stage32(const bf16* g, bf16* l) {
#if HAS_ASYNC_LDS
  __builtin_amdgcn_global_load_async_to_lds_b128((v4i*)g, (lds_v4i*)l, 0, 0);
  __builtin_amdgcn_global_load_async_to_lds_b128((v4i*)g, (lds_v4i*)l, 16, 0);
#else
  *(v8bf*)l       = *(const v8bf*)g;
  *((v8bf*)l + 1) = *((const v8bf*)g + 1);
#endif
}
__device__ __forceinline__ void stage_drain() {
#if HAS_ASYNC_LDS
  asm volatile("s_wait_asynccnt 0" ::: "memory");
#endif
}

// ------------------------------------------------ transpose-convert f32[K][N] -> bf16[N][K]
__global__ void k_convT(const float* __restrict__ in, bf16* __restrict__ out, int K, int N) {
  size_t total = (size_t)K * N;
  size_t stride = (size_t)gridDim.x * blockDim.x;
  for (size_t i = (size_t)blockIdx.x * blockDim.x + threadIdx.x; i < total; i += stride) {
    int k = (int)(i / N), n = (int)(i % N);
    out[(size_t)n * K + k] = f2bf(in[i]);
  }
}

// ------------------------------------------------ embedding (residual stream stays fp32)
__global__ void k_embed(const int* __restrict__ idx, const float* __restrict__ tok,
                        const float* __restrict__ pos, float* __restrict__ x) {
  int bt = blockIdx.x;
  int t  = bt % T_;
  int row = idx[bt];
  const float4* tw = (const float4*)(tok + (size_t)row * E_);
  const float4* pw = (const float4*)(pos + (size_t)t * E_);
  float4* xo = (float4*)(x + (size_t)bt * E_);
  int j = threadIdx.x;                        // 256 threads, E/4 = 256 float4
  float4 a = tw[j], b = pw[j];
  xo[j] = make_float4(a.x + b.x, a.y + b.y, a.z + b.z, a.w + b.w);
}

// ------------------------------------------------ layernorm (fp32 in -> bf16 out, feeds WMMA)
__global__ void k_layernorm(const float* __restrict__ x, const float* __restrict__ g,
                            const float* __restrict__ b, bf16* __restrict__ out) {
  __shared__ float s1[256], s2[256];
  int row = blockIdx.x, tid = threadIdx.x;
  const float* xr = x + (size_t)row * E_;
  float vals[4], sum = 0.f, sq = 0.f;
#pragma unroll
  for (int j = 0; j < 4; ++j) {
    float v = xr[tid + j * 256];
    vals[j] = v; sum += v; sq += v * v;
  }
  s1[tid] = sum; s2[tid] = sq;
  __syncthreads();
  for (int o = 128; o > 0; o >>= 1) {
    if (tid < o) { s1[tid] += s1[tid + o]; s2[tid] += s2[tid + o]; }
    __syncthreads();
  }
  float mean = s1[0] * (1.0f / E_);
  float var  = s2[0] * (1.0f / E_) - mean * mean;
  float rstd = rsqrtf(var + 1e-5f);
#pragma unroll
  for (int j = 0; j < 4; ++j) {
    int c = tid + j * 256;
    out[(size_t)row * E_ + c] = f2bf((vals[j] - mean) * rstd * g[c] + b[c]);
  }
}

// ------------------------------------------------ GEMM (bf16 WMMA, f32 accum)
// C[M,N] = A[M,K](bf16) x BT[N,K](bf16, pre-transposed). Double-buffered LDS,
// async global->LDS staging, block tile 128x128, 8 wave32s, wave tile 32x64.
// modes: 0: acc+bias -> outF (f32)      1: resid+acc+bias -> outF (f32)
//        2: gelu(acc+bias) -> outQ (bf16, row-major)
//        3: bf16 acc -> outQ[b,h,t,d]   4: bf16 acc -> outQ[b,h,d,t] (V for P.V)
__global__ __launch_bounds__(256) void k_gemm(
    const bf16* __restrict__ A, const bf16* __restrict__ BT,
    const float* __restrict__ bias, const float* __restrict__ resid,
    float* __restrict__ outF, bf16* __restrict__ outQ,
    int M, int N, int K, int mode) {
  __shared__ bf16 As[2][128][40];     // [buf][m][k] 128x32 + pad (80B rows, 16B aligned)
  __shared__ bf16 BsT[2][128][40];    // [buf][n][k]

  int tid = threadIdx.x;
  int lane = tid & 31, wid = tid >> 5;
  int lane16 = lane & 15, hi = (lane >> 4) & 1;
  int wm = wid >> 1, wn = wid & 1;                 // 4x2 wave grid
  int bm = blockIdx.y * 128, bn = blockIdx.x * 128;

  v8f acc[2][4];
#pragma unroll
  for (int s = 0; s < 2; ++s)
#pragma unroll
    for (int i = 0; i < 4; ++i) acc[s][i] = (v8f){0.f,0.f,0.f,0.f,0.f,0.f,0.f,0.f};

  int ksteps = K >> 5;
  int srow = tid >> 1, sc0 = (tid & 1) << 4;       // staging: 16 elems/thread, 128 rows
  const bf16* agBase = A  + (size_t)(bm + srow) * K + sc0;
  const bf16* bgBase = BT + (size_t)(bn + srow) * K + sc0;

  auto stage = [&](int kt, int buf) {
    stage32(agBase + (kt << 5), &As[buf][srow][sc0]);
    stage32(bgBase + (kt << 5), &BsT[buf][srow][sc0]);
  };

  stage(0, 0);
  stage_drain();
  __syncthreads();

  for (int kt = 0; kt < ksteps; ++kt) {
    int cur = kt & 1;
    if (kt + 1 < ksteps) {
      stage(kt + 1, cur ^ 1);                      // DMA overlaps WMMA below
      if (kt + 2 < ksteps) {                       // gfx1250 global_prefetch_b8
        __builtin_prefetch(agBase + ((kt + 2) << 5), 0, 1);
        __builtin_prefetch(bgBase + ((kt + 2) << 5), 0, 1);
      }
    }
    // 4 B fragments (reused by both M sub-tiles): 2x ds_load_b128 each
    v16bf bfrag[4];
#pragma unroll
    for (int nt = 0; nt < 4; ++nt) {
      const bf16* bp = &BsT[cur][wn * 64 + nt * 16 + lane16][hi * 8];
      bfrag[nt] = ldfrag(bp, bp + 16);
    }
#pragma unroll
    for (int sub = 0; sub < 2; ++sub) {
      const bf16* ap = &As[cur][wm * 32 + sub * 16 + lane16][hi * 8];
      v16bf af = ldfrag(ap, ap + 16);   // K halves at +0 / +16 per ISA A layout
#pragma unroll
      for (int nt = 0; nt < 4; ++nt)
        acc[sub][nt] = __builtin_amdgcn_wmma_f32_16x16x32_bf16(
            false, af, false, bfrag[nt], (short)0, acc[sub][nt], false, false);
    }
    stage_drain();                                 // ASYNCcnt -> 0 before publishing tile
    __syncthreads();
  }

  // fused epilogue
#pragma unroll
  for (int sub = 0; sub < 2; ++sub) {
#pragma unroll
    for (int nt = 0; nt < 4; ++nt) {
      int gcol = bn + wn * 64 + nt * 16 + lane16;
#pragma unroll
      for (int r = 0; r < 8; ++r) {
        int grow = bm + wm * 32 + sub * 16 + hi * 8 + r;
        float v = acc[sub][nt][r];
        if (bias) v += bias[gcol];
        if (mode == 1) v += resid[(size_t)grow * N + gcol];
        else if (mode == 2) v = 0.5f * v * (1.0f + erff(v * 0.70710678118654752f));
        if (mode == 2) {
          outQ[(size_t)grow * N + gcol] = f2bf(v);
        } else if (mode == 3) {
          int b = grow / T_, t = grow % T_;
          int h = gcol >> 6, d = gcol & 63;
          outQ[(((size_t)(b * H_ + h)) * T_ + t) * D_ + d] = f2bf(v);
        } else if (mode == 4) {
          int b = grow / T_, t = grow % T_;
          int h = gcol >> 6, d = gcol & 63;
          outQ[(((size_t)(b * H_ + h)) * D_ + d) * T_ + t] = f2bf(v);
        } else {
          outF[(size_t)grow * N + gcol] = v;
        }
      }
    }
  }
}

// ------------------------------------------------ fused causal attention
// One block = 16 query rows of one (b,h). S (transposed) + P live in LDS.
// q,k: [b,h,t,d] bf16 ; v: [b,h,d,t] bf16 ; output y is bf16 (feeds proj GEMM).
__global__ __launch_bounds__(128) void k_attention(
    const bf16* __restrict__ qb, const bf16* __restrict__ kb,
    const bf16* __restrict__ vbT, bf16* __restrict__ y) {
  extern __shared__ char smem[];
  float* S  = (float*)smem;               // [1024][16] f32 (transposed scores)
  bf16*  P  = (bf16*)(S + 16 * T_);       // [16][1024] bf16
  float* rowsum = (float*)(P + 16 * T_);  // 16 f32

  int tid = threadIdx.x;
  int lane = tid & 31, wid = tid >> 5;
  int lane16 = lane & 15, hi = (lane >> 4) & 1;
  int mtile = blockIdx.x, bh = blockIdx.y;
  int b = bh / H_, h = bh % H_;
  const bf16* qbh = qb  + (size_t)bh * T_ * D_;
  const bf16* kbh = kb  + (size_t)bh * T_ * D_;
  const bf16* vbh = vbT + (size_t)bh * D_ * T_;
  const float scale = 0.125f;             // 1/sqrt(64)

  // Q fragments for both 32-wide K-steps over D=64 (two b128 loads each)
  v16bf qa[2];
  int qrow = mtile * 16 + lane16;
#pragma unroll
  for (int ks = 0; ks < 2; ++ks) {
    const bf16* qp = qbh + (size_t)qrow * D_ + ks * 32 + hi * 8;
    qa[ks] = ldfrag(qp, qp + 16);
  }

  // scores: causality prunes key tiles > mtile (wave-uniform -> EXEC all ones)
  for (int nt = wid; nt <= mtile; nt += 4) {
    v8f sacc = (v8f){0.f,0.f,0.f,0.f,0.f,0.f,0.f,0.f};
    int tkey = nt * 16 + lane16;
#pragma unroll
    for (int ks = 0; ks < 2; ++ks) {
      const bf16* kp = kbh + (size_t)tkey * D_ + ks * 32 + hi * 8;
      v16bf kf = ldfrag(kp, kp + 16);     // B = K^T: contraction d, col = key
      sacc = __builtin_amdgcn_wmma_f32_16x16x32_bf16(
          false, qa[ks], false, kf, (short)0, sacc, false, false);
    }
    // transposed store: 8 consecutive floats -> two float4 LDS stores
    float tmp[8];
#pragma unroll
    for (int r = 0; r < 8; ++r) {
      int ql = hi * 8 + r;
      float v = sacc[r] * scale;
      if (tkey > mtile * 16 + ql) v = -1e30f;
      tmp[r] = v;
    }
    float* sp = &S[tkey * 16 + hi * 8];
    *(float4*)sp       = make_float4(tmp[0], tmp[1], tmp[2], tmp[3]);
    *(float4*)(sp + 4) = make_float4(tmp[4], tmp[5], tmp[6], tmp[7]);
  }
  __syncthreads();

  // softmax: 8 threads per row (rows 0..15)
  int r = tid >> 3, le = tid & 7;
  int qidx = mtile * 16 + r;
  int nsteps = (mtile >> 1) + 1;
  int kpad = nsteps * 32;
  float mx = -1e30f;
  for (int s = le; s <= qidx; s += 8) mx = fmaxf(mx, S[s * 16 + r]);
  for (int o = 4; o > 0; o >>= 1) mx = fmaxf(mx, __shfl_xor(mx, o, 8));
  float sum = 0.f;
  for (int s = le; s < kpad; s += 8) {
    float v = (s <= qidx) ? __expf(S[s * 16 + r] - mx) : 0.f;
    P[r * T_ + s] = f2bf(v);
    sum += v;
  }
  for (int o = 4; o > 0; o >>= 1) sum += __shfl_xor(sum, o, 8);
  if (le == 0) rowsum[r] = sum;
  __syncthreads();

  // P x V : each wave owns one 16-wide d-tile, loops all key steps
  v8f yacc = (v8f){0.f,0.f,0.f,0.f,0.f,0.f,0.f,0.f};
  int d = wid * 16 + lane16;
  for (int ks = 0; ks < nsteps; ++ks) {
    int kb0 = ks * 32;
    const bf16* pp = &P[lane16 * T_ + kb0 + hi * 8];
    v16bf pa = ldfrag(pp, pp + 16);
    const bf16* vp = vbh + (size_t)d * T_ + kb0 + hi * 8;  // [d][t]: contiguous in t
    v16bf vf = ldfrag(vp, vp + 16);
    yacc = __builtin_amdgcn_wmma_f32_16x16x32_bf16(
        false, pa, false, vf, (short)0, yacc, false, false);
  }
#pragma unroll
  for (int rr = 0; rr < 8; ++rr) {
    int ql = hi * 8 + rr;
    int t = mtile * 16 + ql;
    y[((size_t)(b * T_ + t)) * E_ + h * D_ + d] = f2bf(yacc[rr] / rowsum[ql]);
  }
}

// ------------------------------------------------ host
extern "C" void kernel_launch(void* const* d_in, const int* in_sizes, int n_in,
                              void* d_out, int out_size, void* d_ws, size_t ws_size,
                              hipStream_t stream) {
  (void)in_sizes; (void)n_in; (void)out_size; (void)ws_size;
  const int*   idx   = (const int*)d_in[0];
  const float* tok_w = (const float*)d_in[1];
  const float* pos_w = (const float*)d_in[2];
  const float* ln1_g = (const float*)d_in[3];
  const float* ln1_b = (const float*)d_in[4];
  const float* wq    = (const float*)d_in[5];
  const float* wk    = (const float*)d_in[6];
  const float* wv    = (const float*)d_in[7];
  const float* wp    = (const float*)d_in[8];
  const float* bp    = (const float*)d_in[9];
  const float* ln2_g = (const float*)d_in[10];
  const float* ln2_b = (const float*)d_in[11];
  const float* w1    = (const float*)d_in[12];
  const float* b1    = (const float*)d_in[13];
  const float* w2    = (const float*)d_in[14];
  const float* b2    = (const float*)d_in[15];
  const float* lnf_g = (const float*)d_in[16];
  const float* lnf_b = (const float*)d_in[17];
  const float* wlm   = (const float*)d_in[18];
  const float* blm   = (const float*)d_in[19];

  char* ws = (char*)d_ws;
  size_t off = 0;
  auto take = [&](size_t bytes) -> char* {
    off = (off + 255) & ~(size_t)255;
    char* p = ws + off;
    off += bytes;
    return p;
  };
  const size_t nE2 = (size_t)E_ * E_;          // 1M
  const size_t nEF = (size_t)E_ * F_;          // 4M
  const size_t nBTE = (size_t)B_ * T_ * E_;    // 2M
  const size_t nBTF = (size_t)B_ * T_ * F_;    // 8M

  bf16* wq_b  = (bf16*)take(2 * L_ * nE2);     // all stored transposed [N][K]
  bf16* wk_b  = (bf16*)take(2 * L_ * nE2);
  bf16* wv_b  = (bf16*)take(2 * L_ * nE2);
  bf16* wp_b  = (bf16*)take(2 * L_ * nE2);
  bf16* w1_b  = (bf16*)take(2 * L_ * nEF);
  bf16* w2_b  = (bf16*)take(2 * L_ * nEF);
  bf16* wlm_b = (bf16*)take(2 * (size_t)E_ * V_);
  float* X   = (float*)take(4 * nBTE);         // fp32 residual stream
  bf16*  Hb  = (bf16*)take(2 * nBTE);          // layernorm out (GEMM A operand)
  bf16*  Yb  = (bf16*)take(2 * nBTE);          // attention out
  bf16*  FFb = (bf16*)take(2 * nBTF);          // gelu out
  bf16* QB = (bf16*)take(2 * nBTE);            // [b,h,t,d]
  bf16* KB = (bf16*)take(2 * nBTE);            // [b,h,t,d]
  bf16* VB = (bf16*)take(2 * nBTE);            // [b,h,d,t]

  auto convT = [&](const float* src, bf16* dst, int K, int N) {
    k_convT<<<4096, 256, 0, stream>>>(src, dst, K, N);
  };
  for (int l = 0; l < L_; ++l) {
    convT(wq + l * nE2, wq_b + l * nE2, E_, E_);
    convT(wk + l * nE2, wk_b + l * nE2, E_, E_);
    convT(wv + l * nE2, wv_b + l * nE2, E_, E_);
    convT(wp + l * nE2, wp_b + l * nE2, E_, E_);
    convT(w1 + l * nEF, w1_b + l * nEF, E_, F_);
    convT(w2 + l * nEF, w2_b + l * nEF, F_, E_);
  }
  convT(wlm, wlm_b, E_, V_);

  auto gemm = [&](const bf16* A, const bf16* BT, const float* bias, const float* resid,
                  float* outF, bf16* outQ, int M, int N, int K, int mode) {
    dim3 g(N / 128, M / 128);
    k_gemm<<<g, 256, 0, stream>>>(A, BT, bias, resid, outF, outQ, M, N, K, mode);
  };

  const int M = B_ * T_;  // 2048
  k_embed<<<B_ * T_, 256, 0, stream>>>(idx, tok_w, pos_w, X);

  const size_t attn_smem = (size_t)16 * T_ * 4 + (size_t)16 * T_ * 2 + 64;
  for (int l = 0; l < L_; ++l) {
    k_layernorm<<<M, 256, 0, stream>>>(X, ln1_g + (size_t)l * E_, ln1_b + (size_t)l * E_, Hb);
    gemm(Hb, wq_b + (size_t)l * nE2, nullptr, nullptr, nullptr, QB, M, E_, E_, 3);
    gemm(Hb, wk_b + (size_t)l * nE2, nullptr, nullptr, nullptr, KB, M, E_, E_, 3);
    gemm(Hb, wv_b + (size_t)l * nE2, nullptr, nullptr, nullptr, VB, M, E_, E_, 4);
    k_attention<<<dim3(T_ / 16, B_ * H_), 128, attn_smem, stream>>>(QB, KB, VB, Yb);
    gemm(Yb, wp_b + (size_t)l * nE2, bp + (size_t)l * E_, X, X, nullptr, M, E_, E_, 1);
    k_layernorm<<<M, 256, 0, stream>>>(X, ln2_g + (size_t)l * E_, ln2_b + (size_t)l * E_, Hb);
    gemm(Hb, w1_b + (size_t)l * nEF, b1 + (size_t)l * F_, nullptr, nullptr, FFb, M, F_, E_, 2);
    gemm(FFb, w2_b + (size_t)l * nEF, b2 + (size_t)l * E_, X, X, nullptr, M, E_, F_, 1);
  }
  k_layernorm<<<M, 256, 0, stream>>>(X, lnf_g, lnf_b, Hb);
  gemm(Hb, wlm_b, blm, nullptr, (float*)d_out, nullptr, M, V_, E_, 0);
}
